// RBFLayer_60361470378557
// MI455X (gfx1250) — matrix operand order repaired and checked by reading, not compiled, not run on hardware
//
#include <hip/hip_runtime.h>

// RBF layer: out = softmax(-GAMMA * ||x - w||^2) over 4096 centers.
// x: (16384,128) f32, W: (4096,128) f32, out: (16384,4096) f32.
// Strategy: bf16 WMMA GEMM for x.W^T, fused exp + row-sum, 16-row x 4096-col
// exp tile staged in 128KB of CDNA5's 320KB LDS (f16), normalized + streamed
// out once (output-BW bound: 256MB @ 23.3TB/s ~= 11us floor).

#define GAMMA 0.05f
constexpr int BATCH   = 16384;
constexpr int CENTERS = 4096;
constexpr int DIM     = 128;   // K
constexpr int NCHUNKS = DIM / 32;  // 4 K-chunks of 32 per WMMA

typedef __attribute__((ext_vector_type(16))) __bf16   v16bf;
typedef __attribute__((ext_vector_type(8)))  float    v8f;
typedef __attribute__((ext_vector_type(4)))  _Float16 v4h;

union Frag32B { uint4 u[2]; v16bf bf; };
union Pack16B { __bf16 b[8]; uint4 u; };

// ---------------------------------------------------------------------------
// Pre-pass 1: convert f32 rows -> bf16, swizzled directly into WMMA fragment
// order so the main kernel's loads are coalesced global_load_b128.
// Layout: [tile][chunk][part(0: V0-3, 1: V4-7)][lane] * 16 bytes.
//   A (x): lane l, h=l>>4: part0 K = 32c + 8h + {0..7}, part1 K = 32c+16+8h+{0..7}
//   B (W): lane l, h=l>>4: part0 K = 32c + 8h + {0..7}, part1 K = 32c+16+8h+{0..7}
// ---------------------------------------------------------------------------
__global__ void pack_frag(const float* __restrict__ src, __bf16* __restrict__ dst,
                          int ntiles, int isB)
{
    int gtid  = blockIdx.x * blockDim.x + threadIdx.x;
    int wave  = gtid >> 5;
    int lane  = gtid & 31;
    int tile  = wave >> 2;   // / NCHUNKS
    int chunk = wave & 3;    // % NCHUNKS
    if (tile >= ntiles) return;
    int row = tile * 16 + (lane & 15);
    int h   = lane >> 4;
#pragma unroll
    for (int part = 0; part < 2; ++part) {
        int ks = isB ? (chunk * 32 +  8 * h + 16 * part)
                     : (chunk * 32 +  8 * h + 16 * part);
        const float* s = src + (size_t)row * DIM + ks;
        float4 f0 = *(const float4*)s;
        float4 f1 = *(const float4*)(s + 4);
        Pack16B pk;
        pk.b[0] = (__bf16)f0.x; pk.b[1] = (__bf16)f0.y;
        pk.b[2] = (__bf16)f0.z; pk.b[3] = (__bf16)f0.w;
        pk.b[4] = (__bf16)f1.x; pk.b[5] = (__bf16)f1.y;
        pk.b[6] = (__bf16)f1.z; pk.b[7] = (__bf16)f1.w;
        uint4* d = (uint4*)dst + ((((size_t)tile * NCHUNKS + chunk) * 2 + part) * 32 + lane);
        *d = pk.u;
    }
}

// Pre-pass 2: dst[r] = -GAMMA * ||src[r,:]||^2
__global__ void neg_gamma_sqnorm(const float* __restrict__ src, float* __restrict__ dst,
                                 int nrows)
{
    int r = blockIdx.x * blockDim.x + threadIdx.x;
    if (r >= nrows) return;
    const float4* p = (const float4*)(src + (size_t)r * DIM);
    float s = 0.0f;
#pragma unroll
    for (int i = 0; i < DIM / 4; ++i) {
        float4 v = p[i];
        s = fmaf(v.x, v.x, s); s = fmaf(v.y, v.y, s);
        s = fmaf(v.z, v.z, s); s = fmaf(v.w, v.w, s);
    }
    dst[r] = -GAMMA * s;
}

// ---------------------------------------------------------------------------
// Main kernel: block = 256 threads (8 waves) handles 16 rows x 4096 centers.
// Wave w does column tiles ct = w, w+8, ..., w+248.
// ---------------------------------------------------------------------------
__launch_bounds__(256)
__global__ void rbf_softmax(const __bf16* __restrict__ xb, const __bf16* __restrict__ wb,
                            const float* __restrict__ sx, const float* __restrict__ sw,
                            float* __restrict__ out)
{
    extern __shared__ unsigned char smem[];
    _Float16* expTile = (_Float16*)smem;                       // 16*4096 f16 = 128KB
    float*    rowSum  = (float*)(smem + 16 * CENTERS * 2);     // 16
    float*    rowRcp  = rowSum + 16;                           // 16

    const int tid   = threadIdx.x;
    const int lane  = tid & 31;
    const int h     = lane >> 4;
    const int nl    = lane & 15;   // C/D layout: N = lane & 15
    const int wv    = tid >> 5;    // wave id 0..7
    const int rtile = blockIdx.x;  // 16-row tile

    if (tid < 16) rowSum[tid] = 0.0f;
    __syncthreads();

    // Load A fragments (same 16 x-rows for all waves), fragment-ordered.
    v16bf a[NCHUNKS];
#pragma unroll
    for (int c = 0; c < NCHUNKS; ++c) {
        Frag32B f;
        const uint4* p = (const uint4*)xb + (((size_t)rtile * NCHUNKS + c) * 2) * 32 + lane;
        f.u[0] = p[0];
        f.u[1] = p[32];
        a[c] = f.bf;
    }

    // -gamma*|x|^2 for this lane's output rows m = v + 8h (M lives on VGPR idx).
    float sxl = sx[rtile * 16 + nl];
    float sxv[8];
#pragma unroll
    for (int v = 0; v < 8; ++v) sxv[v] = __shfl(sxl, v + 8 * h, 32);

    float rsum[8];
#pragma unroll
    for (int v = 0; v < 8; ++v) rsum[v] = 0.0f;

    for (int ct = wv; ct < CENTERS / 16; ct += 8) {
        float swl = sw[ct * 16 + nl];   // -gamma*|w|^2, N on lanes: matches C/D
        v8f acc = {0.f, 0.f, 0.f, 0.f, 0.f, 0.f, 0.f, 0.f};
#pragma unroll
        for (int c = 0; c < NCHUNKS; ++c) {
            Frag32B f;
            const uint4* p = (const uint4*)wb + (((size_t)ct * NCHUNKS + c) * 2) * 32 + lane;
            f.u[0] = p[0];
            f.u[1] = p[32];
            acc = __builtin_amdgcn_wmma_f32_16x16x32_bf16(
                /*neg_a=*/false, a[c], /*neg_b=*/false, f.bf,
                /*c_mod=*/(short)0, acc, /*reuse_a=*/false, /*reuse_b=*/false);
        }
#pragma unroll
        for (int v = 0; v < 8; ++v) {
            // exp(-g*l2) = exp(2g*dot - g|x|^2 - g|w|^2); l2>=0 => p in (0,1]
            float t = __builtin_fmaf(2.0f * GAMMA, acc[v], sxv[v] + swl);
            float p = __expf(t);
            rsum[v] += p;
            expTile[(v + 8 * h) * CENTERS + ct * 16 + nl] = (_Float16)p;
        }
    }

    // Reduce row sums over the 16 lanes of each half-wave (N direction).
#pragma unroll
    for (int v = 0; v < 8; ++v) {
        float s = rsum[v];
        s += __shfl_xor(s, 1, 32);
        s += __shfl_xor(s, 2, 32);
        s += __shfl_xor(s, 4, 32);
        s += __shfl_xor(s, 8, 32);
        rsum[v] = s;
    }
    if (nl == 0) {
#pragma unroll
        for (int v = 0; v < 8; ++v)
            atomicAdd(&rowSum[v + 8 * h], rsum[v]);   // ds_add_f32
    }
    __syncthreads();

    if (tid < 16) rowRcp[tid] = 1.0f / rowSum[tid];
    __syncthreads();

    // Phase 2: normalize + stream out, fully coalesced float4 stores.
    const size_t outBase = (size_t)rtile * 16 * CENTERS;
#pragma unroll 4
    for (int i = 0; i < (16 * CENTERS / 4) / 256; ++i) {   // 64 iterations
        int f  = i * 256 + tid;       // float4 index within the 16x4096 tile
        int m  = f >> 10;             // 1024 float4 per row
        int c4 = f & 1023;
        v4h hv = *(const v4h*)(expTile + (m * CENTERS + c4 * 4));
        float r = rowRcp[m];
        float4 o;
        o.x = (float)hv.x * r;
        o.y = (float)hv.y * r;
        o.z = (float)hv.z * r;
        o.w = (float)hv.w * r;
        *(float4*)(out + outBase + (size_t)f * 4) = o;
    }
}

// ---------------------------------------------------------------------------
extern "C" void kernel_launch(void* const* d_in, const int* in_sizes, int n_in,
                              void* d_out, int out_size, void* d_ws, size_t ws_size,
                              hipStream_t stream)
{
    (void)in_sizes; (void)n_in; (void)out_size; (void)ws_size;
    const float* x = (const float*)d_in[0];
    const float* W = (const float*)d_in[1];
    float* out = (float*)d_out;

    unsigned char* ws = (unsigned char*)d_ws;
    __bf16* xb = (__bf16*)ws;                                   // 4 MB
    __bf16* wb = (__bf16*)(ws + (size_t)BATCH * DIM * 2);       // 1 MB
    float*  sx = (float*)(ws + (size_t)BATCH * DIM * 2 + (size_t)CENTERS * DIM * 2);
    float*  sw = sx + BATCH;

    // Pre-pass: bf16 fragment packing + (-gamma * squared norms).
    {
        int threads = (BATCH / 16) * NCHUNKS * 32;   // one wave per (tile,chunk)
        pack_frag<<<threads / 256, 256, 0, stream>>>(x, xb, BATCH / 16, 0);
    }
    {
        int threads = (CENTERS / 16) * NCHUNKS * 32;
        pack_frag<<<threads / 256, 256, 0, stream>>>(W, wb, CENTERS / 16, 1);
    }
    neg_gamma_sqnorm<<<BATCH / 256, 256, 0, stream>>>(x, sx, BATCH);
    neg_gamma_sqnorm<<<CENTERS / 256, 256, 0, stream>>>(W, sw, CENTERS);

    // Main fused GEMM + softmax. Dynamic LDS: 128KB exp tile + 2x16 floats.
    size_t shmem = (size_t)16 * CENTERS * 2 + 32 * sizeof(float);
    rbf_softmax<<<BATCH / 16, 256, shmem, stream>>>(xb, wb, sx, sw, out);
}
